// CrossModalAttention_81252191306396
// MI455X (gfx1250) — matrix operand order repaired and checked by reading, not compiled, not run on hardware
//
#include <hip/hip_runtime.h>
#include <hip/hip_bf16.h>
#include <stdint.h>

typedef __bf16 bf16;
typedef __attribute__((ext_vector_type(16))) __bf16 v16bf;
typedef __attribute__((ext_vector_type(8)))  __bf16 v8bf;
typedef __attribute__((ext_vector_type(8)))  float  v8f;
typedef __attribute__((__vector_size__(16))) int    v4i;

#define NB_B 8
#define NB_S 1024
#define NB_D 768
#define NB_H 8
#define NB_DH 96
#define ROWS 16384      // 2*B*S rows: [visual ; text]
#define HALF_ROWS 8192
static constexpr float INV_SQRT_DH = 0.10206207261596575f; // 1/sqrt(96)

#if defined(__AMDGCN__) && __has_builtin(__builtin_amdgcn_global_load_async_to_lds_b128) && \
    __has_builtin(__builtin_amdgcn_s_wait_asynccnt)
#define HAS_ASYNC_LDS 1
#else
#define HAS_ASYNC_LDS 0
#endif

__device__ __forceinline__ v16bf cat8(v8bf a, v8bf b) {
  return __builtin_shufflevector(a, b, 0,1,2,3,4,5,6,7,8,9,10,11,12,13,14,15);
}
__device__ __forceinline__ v8f wmma_bf16(v16bf a, v16bf b, v8f c) {
  return __builtin_amdgcn_wmma_f32_16x16x32_bf16(false, a, false, b, (short)0, c,
                                                 false, false);
}

// 16-lane (DPP row) reductions via row_ror rotations: every lane of the
// 16-lane group ends with the full reduction. Pure VALU, no LDS traffic.
__device__ __forceinline__ float row16_max(float x) {
#if defined(__AMDGCN__) && __has_builtin(__builtin_amdgcn_update_dpp)
  x = fmaxf(x, __builtin_bit_cast(float, __builtin_amdgcn_update_dpp(
                   0, __builtin_bit_cast(int, x), 0x121, 0xf, 0xf, true)));
  x = fmaxf(x, __builtin_bit_cast(float, __builtin_amdgcn_update_dpp(
                   0, __builtin_bit_cast(int, x), 0x122, 0xf, 0xf, true)));
  x = fmaxf(x, __builtin_bit_cast(float, __builtin_amdgcn_update_dpp(
                   0, __builtin_bit_cast(int, x), 0x124, 0xf, 0xf, true)));
  x = fmaxf(x, __builtin_bit_cast(float, __builtin_amdgcn_update_dpp(
                   0, __builtin_bit_cast(int, x), 0x128, 0xf, 0xf, true)));
#else
  x = fmaxf(x, __shfl_xor(x, 1, 32));
  x = fmaxf(x, __shfl_xor(x, 2, 32));
  x = fmaxf(x, __shfl_xor(x, 4, 32));
  x = fmaxf(x, __shfl_xor(x, 8, 32));
#endif
  return x;
}
__device__ __forceinline__ float row16_sum(float x) {
#if defined(__AMDGCN__) && __has_builtin(__builtin_amdgcn_update_dpp)
  x += __builtin_bit_cast(float, __builtin_amdgcn_update_dpp(
           0, __builtin_bit_cast(int, x), 0x121, 0xf, 0xf, true));
  x += __builtin_bit_cast(float, __builtin_amdgcn_update_dpp(
           0, __builtin_bit_cast(int, x), 0x122, 0xf, 0xf, true));
  x += __builtin_bit_cast(float, __builtin_amdgcn_update_dpp(
           0, __builtin_bit_cast(int, x), 0x124, 0xf, 0xf, true));
  x += __builtin_bit_cast(float, __builtin_amdgcn_update_dpp(
           0, __builtin_bit_cast(int, x), 0x128, 0xf, 0xf, true));
#else
  x += __shfl_xor(x, 1, 32);
  x += __shfl_xor(x, 2, 32);
  x += __shfl_xor(x, 4, 32);
  x += __shfl_xor(x, 8, 32);
#endif
  return x;
}

// ---------------------------------------------------------------------------
// 1) f32 -> bf16 convert of concatenated activations X = [visual ; text]
// ---------------------------------------------------------------------------
__global__ __launch_bounds__(256) void k_tobf16_x(const float* __restrict__ vis,
                                                  const float* __restrict__ txt,
                                                  bf16* __restrict__ xbf) {
  size_t i = (size_t)blockIdx.x * 256 + threadIdx.x;
  const size_t half = (size_t)HALF_ROWS * NB_D;
  float v = (i < half) ? vis[i] : txt[i - half];
  xbf[i] = (bf16)v;
}

// ---------------------------------------------------------------------------
// 2) weight f32 [k][n] -> bf16 transposed [n][k]
// ---------------------------------------------------------------------------
__global__ __launch_bounds__(256) void k_tobf16_wt(const float* __restrict__ w,
                                                   bf16* __restrict__ wt) {
  __shared__ float tile[32][33];
  int k0 = blockIdx.y * 32, n0 = blockIdx.x * 32;
  int tx = threadIdx.x & 31, ty = threadIdx.x >> 5;
  for (int r = ty; r < 32; r += 8)
    tile[r][tx] = w[(size_t)(k0 + r) * NB_D + n0 + tx];
  __syncthreads();
  for (int r = ty; r < 32; r += 8)
    wt[(size_t)(n0 + r) * NB_D + k0 + tx] = (bf16)tile[tx][r];
}

// ---------------------------------------------------------------------------
// 3) GEMM: C[M,768] = A(bf16) x W (via Wt[n][k]) + bias, *scale.
//    Block = 8 M-waves, 128(M) x 64(N) tile. Weight tile (64 cols x 32 k)
//    staged once per K-step in LDS and shared by all 8 waves.
// ---------------------------------------------------------------------------
template <bool OUT_F32>
__global__ __launch_bounds__(256) void k_gemm(const bf16* __restrict__ A,
                                              const bf16* __restrict__ Wt,
                                              const float* __restrict__ bias,
                                              float scale,
                                              bf16* __restrict__ Cbf,
                                              float* __restrict__ Cf) {
  const int lane = threadIdx.x & 31;
  const int wave = threadIdx.x >> 5;
  const int ln   = lane & 15;
  const int hi   = lane >> 4;
  const int rowA = blockIdx.y * 128 + wave * 16 + ln;
  const int nB   = blockIdx.x * 64;

  __shared__ bf16 wtile[64][40];   // [n-local][k-local], padded rows (80B)

  v8f acc[4] = {v8f{}, v8f{}, v8f{}, v8f{}};
  const bf16* aRow = A + (size_t)rowA * NB_D;

  for (int k0 = 0; k0 < NB_D; k0 += 32) {
    __syncthreads();
    for (int id = threadIdx.x; id < 64 * 16; id += 256) {
      int rr = id >> 4, cc = id & 15;
      reinterpret_cast<uint32_t*>(&wtile[rr][0])[cc] =
          reinterpret_cast<const uint32_t*>(Wt + (size_t)(nB + rr) * NB_D + k0)[cc];
    }
    __syncthreads();

    // A fragment: lane<16 K={0..7,16..23}, lane>=16 K={8..15,24..31}
    v8bf a0 = *(const v8bf*)(aRow + k0 + hi * 8);
    v8bf a1 = *(const v8bf*)(aRow + k0 + 16 + hi * 8);
    v16bf afrag = cat8(a0, a1);
#pragma unroll
    for (int nc = 0; nc < 4; ++nc) {
      const bf16* wRow = &wtile[nc * 16 + ln][hi * 8];
      v16bf bfrag = cat8(*(const v8bf*)wRow, *(const v8bf*)(wRow + 16));
      acc[nc] = wmma_bf16(afrag, bfrag, acc[nc]);
    }
  }

#pragma unroll
  for (int nc = 0; nc < 4; ++nc) {
    int col = nB + nc * 16 + ln;
    float bv = bias[col];
#pragma unroll
    for (int r = 0; r < 8; ++r) {
      int m = blockIdx.y * 128 + wave * 16 + hi * 8 + r;
      float v = (acc[nc][r] + bv) * scale;
      if (OUT_F32) Cf[(size_t)m * NB_D + col] = v;
      else         Cbf[(size_t)m * NB_D + col] = (bf16)v;
    }
  }
}

// ---------------------------------------------------------------------------
// 3b) V -> V^T in global: Vt[(sb*8+h)][dh][S] so attention can stage V tiles
//     with plain row-wise (async) copies instead of an in-kernel transpose.
// ---------------------------------------------------------------------------
__global__ __launch_bounds__(256) void k_vtrans(const bf16* __restrict__ V,
                                                bf16* __restrict__ Vt) {
  __shared__ bf16 t[32][33];
  int st = blockIdx.x;               // 32 s-tiles
  int dt = blockIdx.y;               // 3 dh-tiles
  int z  = blockIdx.z;               // sb*8 + h, sb in [0,16)
  int h  = z & 7, sb = z >> 3;
  int tx = threadIdx.x & 31, ty = threadIdx.x >> 5;
  const bf16* src = V + ((size_t)sb * NB_S + st * 32) * NB_D + h * NB_DH + dt * 32;
  for (int r = ty; r < 32; r += 8) t[r][tx] = src[(size_t)r * NB_D + tx];
  __syncthreads();
  bf16* dst = Vt + ((size_t)z * NB_DH + dt * 32) * NB_S + st * 32;
  for (int r = ty; r < 32; r += 8) dst[(size_t)r * NB_S + tx] = t[tx][r];
}

// ---------------------------------------------------------------------------
// 4) Flash attention with async double-buffered K/V tiles.
//    grid = (8 q-tiles, H, dir*B); 8 waves x 16 q-rows each.
// ---------------------------------------------------------------------------
__global__ __launch_bounds__(256) void k_attn(const bf16* __restrict__ Qbf,
                                              const bf16* __restrict__ Kbf,
                                              const bf16* __restrict__ VtG,
                                              const int* __restrict__ vmask,
                                              const int* __restrict__ tmask,
                                              bf16* __restrict__ outp) {
  const int qt  = blockIdx.x;
  const int h   = blockIdx.y;
  const int db  = blockIdx.z;
  const int dir = db >> 3;
  const int b   = db & 7;

  const size_t qSeg = dir == 0 ? 0 : (size_t)HALF_ROWS * NB_D;
  const size_t kSeg = dir == 0 ? (size_t)HALF_ROWS * NB_D : 0;
  const int kvsb    = dir == 0 ? (8 + b) : b;             // kv segment-batch id
  const int* mask   = dir == 0 ? tmask : vmask;

  const int lane = threadIdx.x & 31;
  const int wave = threadIdx.x >> 5;
  const int ln   = lane & 15;
  const int hi   = lane >> 4;

  __shared__ bf16  kT[2][32][104];     // [buf][key][dh] padded (208B rows)
  __shared__ bf16  vT[2][96][40];      // [buf][dh][key] padded (80B rows)
  __shared__ float pBuf[8][16][36];    // per-wave P transpose staging

  const bf16* kBase  = Kbf + kSeg + ((size_t)b * NB_S) * NB_D + h * NB_DH;
  const size_t vtRow = ((size_t)(kvsb * NB_H + h)) * NB_DH * NB_S;
  const bf16* vBase  = VtG + vtRow;

  // stage one 32-key tile (K row-major, V^T row-major) into buffer bufN
  auto stage = [&](int kb, int bufN) {
    for (int id = threadIdx.x; id < 384; id += 256) {
      int kr = id / 12, kc = id % 12;                  // K: 32 rows x 12 chunks
      const bf16* gk = kBase + (size_t)(kb + kr) * NB_D + kc * 8;
      bf16* lk = &kT[bufN][kr][kc * 8];
      int vr = id >> 2, vc = id & 3;                   // V: 96 rows x 4 chunks
      const bf16* gv = vBase + (size_t)vr * NB_S + kb + vc * 8;
      bf16* lv = &vT[bufN][vr][vc * 8];
#if HAS_ASYNC_LDS
      __builtin_amdgcn_global_load_async_to_lds_b128((v4i*)gk, (v4i*)lk, 0, 0);
      __builtin_amdgcn_global_load_async_to_lds_b128((v4i*)gv, (v4i*)lv, 0, 0);
#else
      *(uint4*)lk = *(const uint4*)gk;
      *(uint4*)lv = *(const uint4*)gv;
#endif
    }
  };

  // Q fragments (16 rows x 96 dh = 3 A-frags), held in registers throughout.
  const bf16* qRowPtr =
      Qbf + qSeg + ((size_t)(b * NB_S + qt * 128 + wave * 16 + ln)) * NB_D + h * NB_DH;
  v16bf qf[3];
#pragma unroll
  for (int c = 0; c < 3; ++c) {
    v8bf a0 = *(const v8bf*)(qRowPtr + c * 32 + hi * 8);
    v8bf a1 = *(const v8bf*)(qRowPtr + c * 32 + 16 + hi * 8);
    qf[c] = cat8(a0, a1);
  }

  float mrow[8], lrow[8];
  v8f o[6] = {v8f{}, v8f{}, v8f{}, v8f{}, v8f{}, v8f{}};
#pragma unroll
  for (int r = 0; r < 8; ++r) { mrow[r] = -3.0e38f; lrow[r] = 0.f; }

  stage(0, 0);   // prologue prefetch

  for (int kb = 0; kb < NB_S; kb += 32) {
    const int cur = (kb >> 5) & 1;
#if HAS_ASYNC_LDS
    __builtin_amdgcn_s_wait_asynccnt(0);
#endif
    __syncthreads();                         // tile ready + prev reads done
    if (kb + 32 < NB_S) stage(kb + 32, cur ^ 1);   // prefetch next tile

    // --- S = Q K^T for 32 keys: two 16x16 tiles, 3 wmma each ---
    v8f s0 = {}, s1 = {};
#pragma unroll
    for (int c = 0; c < 3; ++c) {
      const bf16* kp0 = &kT[cur][ln][c * 32 + hi * 8];
      s0 = wmma_bf16(qf[c], cat8(*(const v8bf*)kp0, *(const v8bf*)(kp0 + 16)), s0);
      const bf16* kp1 = &kT[cur][16 + ln][c * 32 + hi * 8];
      s1 = wmma_bf16(qf[c], cat8(*(const v8bf*)kp1, *(const v8bf*)(kp1 + 16)), s1);
    }
    float mA0 = mask[b * NB_S + kb + ln]      ? 0.f : -3.0e38f;
    float mA1 = mask[b * NB_S + kb + 16 + ln] ? 0.f : -3.0e38f;

    // --- online softmax: row stats via DPP row-rotate reductions ---
    float nm[8], alpha[8];
    v8f p0, p1;
#pragma unroll
    for (int r = 0; r < 8; ++r) {
      float a = s0[r] + mA0, bb = s1[r] + mA1;
      s0[r] = a; s1[r] = bb;
      float t = row16_max(fmaxf(a, bb));
      nm[r] = fmaxf(mrow[r], t);
      alpha[r] = __expf(mrow[r] - nm[r]);
      mrow[r] = nm[r];
      p0[r] = __expf(s0[r] - nm[r]);
      p1[r] = __expf(s1[r] - nm[r]);
      float rs = row16_sum(p0[r] + p1[r]);
      lrow[r] = lrow[r] * alpha[r] + rs;
    }
#pragma unroll
    for (int n = 0; n < 6; ++n)
#pragma unroll
      for (int r = 0; r < 8; ++r) o[n][r] *= alpha[r];

    // --- transpose P (C/D layout) -> A-frag layout via per-wave LDS ---
    float* pw = &pBuf[wave][0][0];
#pragma unroll
    for (int r = 0; r < 8; ++r) {
      int prow = hi * 8 + r;
      pw[prow * 36 + ln]      = p0[r];
      pw[prow * 36 + 16 + ln] = p1[r];
    }
    const float* pr = pw + ln * 36;          // wave-private: ds ops in-order
    float4 fa = *(const float4*)(pr + hi * 8);
    float4 fb = *(const float4*)(pr + hi * 8 + 4);
    float4 fc = *(const float4*)(pr + 16 + hi * 8);
    float4 fd = *(const float4*)(pr + 16 + hi * 8 + 4);
    v16bf pf;
    pf[0]=(bf16)fa.x; pf[1]=(bf16)fa.y; pf[2]=(bf16)fa.z; pf[3]=(bf16)fa.w;
    pf[4]=(bf16)fb.x; pf[5]=(bf16)fb.y; pf[6]=(bf16)fb.z; pf[7]=(bf16)fb.w;
    pf[8]=(bf16)fc.x; pf[9]=(bf16)fc.y; pf[10]=(bf16)fc.z; pf[11]=(bf16)fc.w;
    pf[12]=(bf16)fd.x; pf[13]=(bf16)fd.y; pf[14]=(bf16)fd.z; pf[15]=(bf16)fd.w;

    // --- O += P V : 6 dh chunks of 16 ---
#pragma unroll
    for (int n = 0; n < 6; ++n) {
      const bf16* vp = &vT[cur][n * 16 + ln][hi * 8];
      o[n] = wmma_bf16(pf, cat8(*(const v8bf*)vp, *(const v8bf*)(vp + 16)), o[n]);
    }
  }

  // --- normalize and store bf16 ---
  float invl[8];
#pragma unroll
  for (int r = 0; r < 8; ++r) invl[r] = lrow[r] > 0.f ? 1.0f / lrow[r] : 0.f;
  int row0 = dir * HALF_ROWS + b * NB_S + qt * 128 + wave * 16 + hi * 8;
#pragma unroll
  for (int n = 0; n < 6; ++n) {
    int col = h * NB_DH + n * 16 + ln;
#pragma unroll
    for (int r = 0; r < 8; ++r)
      outp[(size_t)(row0 + r) * NB_D + col] = (bf16)(o[n][r] * invl[r]);
  }
}

// ---------------------------------------------------------------------------
// 5) residual + LayerNorm: out = LN(x + Ypre) * g + b. One block per row.
// ---------------------------------------------------------------------------
__global__ __launch_bounds__(256) void k_ln(const float* __restrict__ ypre,
                                            const float* __restrict__ vis,
                                            const float* __restrict__ txt,
                                            const float* __restrict__ g,
                                            const float* __restrict__ be,
                                            float* __restrict__ out) {
  int row = blockIdx.x;
  const float* res = (row < HALF_ROWS) ? (vis + (size_t)row * NB_D)
                                       : (txt + (size_t)(row - HALF_ROWS) * NB_D);
  const float* y = ypre + (size_t)row * NB_D;
  float vals[3], s = 0.f, s2 = 0.f;
#pragma unroll
  for (int i = 0; i < 3; ++i) {
    int c = threadIdx.x + i * 256;
    float v = res[c] + y[c];
    vals[i] = v; s += v; s2 += v * v;
  }
#pragma unroll
  for (int off = 16; off >= 1; off >>= 1) {
    s += __shfl_xor(s, off, 32);
    s2 += __shfl_xor(s2, off, 32);
  }
  __shared__ float red[18];
  int wave = threadIdx.x >> 5, lane = threadIdx.x & 31;
  if (lane == 0) { red[wave] = s; red[8 + wave] = s2; }
  __syncthreads();
  if (threadIdx.x == 0) {
    float ts = 0.f, ts2 = 0.f;
    for (int w = 0; w < 8; ++w) { ts += red[w]; ts2 += red[8 + w]; }
    float mean = ts / (float)NB_D;
    red[16] = mean;
    red[17] = ts2 / (float)NB_D - mean * mean;
  }
  __syncthreads();
  float mean = red[16];
  float rstd = rsqrtf(red[17] + 1e-5f);
#pragma unroll
  for (int i = 0; i < 3; ++i) {
    int c = threadIdx.x + i * 256;
    out[(size_t)row * NB_D + c] = (vals[i] - mean) * rstd * g[c] + be[c];
  }
}

// ---------------------------------------------------------------------------
extern "C" void kernel_launch(void* const* d_in, const int* in_sizes, int n_in,
                              void* d_out, int out_size, void* d_ws, size_t ws_size,
                              hipStream_t stream) {
  const float* vis = (const float*)d_in[0];
  const float* txt = (const float*)d_in[1];
  const float* Wq  = (const float*)d_in[2];
  const float* bq  = (const float*)d_in[3];
  const float* Wk  = (const float*)d_in[4];
  const float* bk  = (const float*)d_in[5];
  const float* Wv  = (const float*)d_in[6];
  const float* bv  = (const float*)d_in[7];
  const float* Wo  = (const float*)d_in[8];
  const float* bo  = (const float*)d_in[9];
  const float* g   = (const float*)d_in[10];
  const float* be  = (const float*)d_in[11];
  const int* vmask = (const int*)d_in[12];
  const int* tmask = (const int*)d_in[13];
  float* out = (float*)d_out;

  const size_t XE = (size_t)ROWS * NB_D;
  const size_t WE = (size_t)NB_D * NB_D;
  bf16* Xbf = (bf16*)d_ws;
  bf16* Qbf = Xbf + XE;
  bf16* Kbf = Qbf + XE;
  bf16* Vbf = Kbf + XE;
  bf16* Abf = Vbf + XE;
  bf16* WqT = Abf + XE;
  bf16* WkT = WqT + WE;
  bf16* WvT = WkT + WE;
  bf16* WoT = WvT + WE;
  bf16* VtG  = Xbf;            // aliases Xbf (dead after the V projection)
  float* Ypre = (float*)Qbf;   // aliases Qbf+Kbf (dead after attention)

  k_tobf16_x<<<dim3((unsigned)(XE / 256)), 256, 0, stream>>>(vis, txt, Xbf);
  dim3 wtg(24, 24);
  k_tobf16_wt<<<wtg, 256, 0, stream>>>(Wq, WqT);
  k_tobf16_wt<<<wtg, 256, 0, stream>>>(Wk, WkT);
  k_tobf16_wt<<<wtg, 256, 0, stream>>>(Wv, WvT);
  k_tobf16_wt<<<wtg, 256, 0, stream>>>(Wo, WoT);

  dim3 gg(NB_D / 64, ROWS / 128);
  k_gemm<false><<<gg, 256, 0, stream>>>(Xbf, WqT, bq, INV_SQRT_DH, Qbf, nullptr);
  k_gemm<false><<<gg, 256, 0, stream>>>(Xbf, WkT, bk, 1.0f, Kbf, nullptr);
  k_gemm<false><<<gg, 256, 0, stream>>>(Xbf, WvT, bv, 1.0f, Vbf, nullptr);

  // V -> V^T (overwrites Xbf region, which is dead now)
  k_vtrans<<<dim3(32, 3, 128), 256, 0, stream>>>(Vbf, VtG);

  k_attn<<<dim3(NB_S / 128, NB_H, 2 * NB_B), 256, 0, stream>>>(
      Qbf, Kbf, VtG, vmask, tmask, Abf);

  k_gemm<true><<<gg, 256, 0, stream>>>(Abf, WoT, bo, 1.0f, nullptr, Ypre);

  k_ln<<<dim3(ROWS), 256, 0, stream>>>(Ypre, vis, txt, g, be, out);
}
